// FeatureExtractorLayer_21784074125679
// MI455X (gfx1250) — matrix operand order repaired and checked by reading, not compiled, not run on hardware
//
#include <hip/hip_runtime.h>
#include <hip/hip_bf16.h>
#include <stdint.h>

// ---------------------------------------------------------------------------
// FeatureExtractorLayer for MI455X (gfx1250, wave32).
// All dense projections run through v_wmma_f32_16x16x32_f16 (f32 accumulate).
// Segment-softmax attention runs as 3 scalar per-edge passes with atomics.
//
// Input order (setup_inputs() insertion order, params flattened in insertion
// order): 0 var_learned_f, 1 var_lp_f, 2 con_learned_f, 3 con_lp_f,
// 4 edge_learned_f, 5 lo_costs, 6 hi_costs, 7 def_mm, 8 edge_lp_f_wo_ss,
// 9 edge_index_var(int), 10 edge_index_con(int),
// 11..19 con_conv {Wq,bq,Wk,bk,Wv,bv,We,Wskip,bskip},
// 20..28 var_conv {same},
// 29..40 edge {vW1,vb1,vW2,vb2,cW1,cb1,cW2,cb2,eW1,eb1,eW2,eb2}.
// Output: concat(var_new[NV*32], con_new[NC*32], edge_new[E*32]) fp32.
// ---------------------------------------------------------------------------

#define NVN 200000
#define NCN 100000
#define EN  2000000
#define KP  64          // padded K for 40-wide node/edge features

typedef _Float16 v16h __attribute__((ext_vector_type(16)));
typedef _Float16 h8   __attribute__((ext_vector_type(8)));
typedef float    v8f  __attribute__((ext_vector_type(8)));

__device__ __forceinline__ h8 ldh8(const _Float16* p) { return *(const h8*)p; }

__device__ __forceinline__ v16h mkfrag(h8 lo, h8 hi) {
  return __builtin_shufflevector(lo, hi, 0,1,2,3,4,5,6,7,8,9,10,11,12,13,14,15);
}

__device__ __forceinline__ v8f wmma_f16(v16h a, v16h b, v8f c) {
  // D = A(16x32 f16) * B(32x16 f16) + C(16x16 f32)
  return __builtin_amdgcn_wmma_f32_16x16x32_f16(false, a, false, b, (short)0, c,
                                                false, false);
}

// order-preserving float <-> uint encoding for atomicMax-based segment max
__device__ __forceinline__ unsigned int fenc(float f) {
  unsigned int u = __float_as_uint(f);
  return (u & 0x80000000u) ? ~u : (u | 0x80000000u);
}
__device__ __forceinline__ float fdec(unsigned int x) {
  return (x & 0x80000000u) ? __uint_as_float(x & 0x7FFFFFFFu)
                           : __uint_as_float(~x);
}

// ---------------------------- packing kernels ------------------------------

__global__ void pack_nodes(const float* __restrict__ learned,
                           const float* __restrict__ lp,
                           _Float16* __restrict__ out, int N) {
  int i = blockIdx.x * blockDim.x + threadIdx.x;
  if (i >= N) return;
  _Float16* o = out + (size_t)i * KP;
  const float* r = learned + (size_t)i * 32;
  const float* l = lp + (size_t)i * 8;
#pragma unroll
  for (int j = 0; j < 32; ++j) o[j] = (_Float16)r[j];
#pragma unroll
  for (int j = 0; j < 8; ++j) o[32 + j] = (_Float16)l[j];
#pragma unroll
  for (int j = 40; j < KP; ++j) o[j] = (_Float16)0.f;
}

__global__ void pack_edges(const float* __restrict__ el,
                           const float* __restrict__ lo,
                           const float* __restrict__ hi,
                           const float* __restrict__ dm,
                           const float* __restrict__ elp,
                           _Float16* __restrict__ out, int E) {
  int e = blockIdx.x * blockDim.x + threadIdx.x;
  if (e >= E) return;
  _Float16* o = out + (size_t)e * KP;
  const float* r = el + (size_t)e * 32;
#pragma unroll
  for (int j = 0; j < 32; ++j) o[j] = (_Float16)r[j];
  o[32] = (_Float16)lo[e];
  o[33] = (_Float16)hi[e];
  o[34] = (_Float16)dm[e];
#pragma unroll
  for (int j = 0; j < 5; ++j) o[35 + j] = (_Float16)elp[(size_t)e * 5 + j];
#pragma unroll
  for (int j = 40; j < KP; ++j) o[j] = (_Float16)0.f;
}

// W is [K,32] row-major f32 -> Wt is [32][Kpad] f16 (transposed, zero padded).
// remap==1 applies the edge-MLP K remap: lds cols 0..39=edge, 64..95=vq,
// 96..127=cq, rest zero (matches A-fragment sourcing in edge_mlp).
__global__ void convert_weight(const float* __restrict__ W,
                               _Float16* __restrict__ Wt, int K, int Kpad,
                               int remap) {
  int i = blockIdx.x * blockDim.x + threadIdx.x;
  int total = 32 * Kpad;
  if (i >= total) return;
  int n = i / Kpad, kk = i % Kpad;
  int ks;
  if (remap) {
    if (kk < 40) ks = kk;
    else if (kk < 64) ks = -1;
    else if (kk < 96) ks = 40 + (kk - 64);
    else ks = 72 + (kk - 96);
  } else {
    ks = (kk < K) ? kk : -1;
  }
  Wt[(size_t)n * Kpad + kk] = (ks >= 0) ? (_Float16)W[(size_t)ks * 32 + n]
                                        : (_Float16)0.f;
}

// --------------------------- generic WMMA GEMM -----------------------------
// Y[M,32] = act(X[M, KCH*32] @ Wt^T + bias); one wave per 16-row tile.
template <int KCH, bool RELU, bool OUTH>
__global__ __launch_bounds__(256) void gemm_rows(
    const _Float16* __restrict__ A, int lda, const _Float16* __restrict__ Wt,
    int ldw, const float* __restrict__ bias, float* __restrict__ outF,
    _Float16* __restrict__ outH, int M) {
  int wave = blockIdx.x * (blockDim.x >> 5) + (threadIdx.x >> 5);
  int base = wave * 16;
  if (base >= M) return;
  int l = threadIdx.x & 31;
  int g = l >> 4;        // half-wave group
  int col = l & 15;      // A-row within tile == C/D column
  const _Float16* arow = A + (size_t)(base + col) * lda;
  v8f acc0 = {}, acc1 = {};
#pragma unroll
  for (int c = 0; c < KCH; ++c) {
    int ka = c * 32;
    v16h a = mkfrag(ldh8(arow + ka + g * 16), ldh8(arow + ka + g * 16 + 8));
    const _Float16* w0 = Wt + (size_t)col * ldw + ka + g * 16;
    const _Float16* w1 = Wt + (size_t)(col + 16) * ldw + ka + g * 16;
    v16h b0 = mkfrag(ldh8(w0), ldh8(w0 + 8));
    v16h b1 = mkfrag(ldh8(w1), ldh8(w1 + 8));
    acc0 = wmma_f16(a, b0, acc0);
    acc1 = wmma_f16(a, b1, acc1);
  }
  float bb0 = bias ? bias[col] : 0.f;
  float bb1 = bias ? bias[col + 16] : 0.f;
#pragma unroll
  for (int r = 0; r < 8; ++r) {
    int row = base + r + 8 * g;
    float o0 = acc0[r] + bb0;
    float o1 = acc1[r] + bb1;
    if (RELU) { o0 = fmaxf(o0, 0.f); o1 = fmaxf(o1, 0.f); }
    if (OUTH) {
      outH[(size_t)row * 32 + col] = (_Float16)o0;
      outH[(size_t)row * 32 + col + 16] = (_Float16)o1;
    } else {
      outF[(size_t)row * 32 + col] = o0;
      outF[(size_t)row * 32 + col + 16] = o1;
    }
  }
}

// ------------------------- attention (segment softmax) ---------------------

__global__ void init_node_state(unsigned int* __restrict__ amaxU,
                                float* __restrict__ denom,
                                float* __restrict__ deg, int N) {
  int i = blockIdx.x * blockDim.x + threadIdx.x;
  if (i >= N) return;
  amaxU[i] = 0x007FFFFFu;  // fenc(-inf)
  denom[i] = 0.f;
  deg[i] = 0.f;
}

__global__ void zero_f32(float* __restrict__ p, int n) {
  int i = blockIdx.x * blockDim.x + threadIdx.x;
  if (i < n) p[i] = 0.f;
}

__global__ void attn_pass1(const int* __restrict__ src,
                           const int* __restrict__ dst,
                           const float* __restrict__ q,
                           const float* __restrict__ k,
                           const _Float16* __restrict__ ep,
                           float* __restrict__ alpha,
                           unsigned int* __restrict__ amaxU, int E) {
  int e = blockIdx.x * blockDim.x + threadIdx.x;
  if (e >= E) return;
  int s = src[e], d = dst[e];
  const float4* q4 = (const float4*)(q + (size_t)d * 32);
  const float4* k4 = (const float4*)(k + (size_t)s * 32);
  const h8* e8 = (const h8*)(ep + (size_t)e * 32);
  float sum = 0.f;
#pragma unroll
  for (int j = 0; j < 4; ++j) {
    float4 qa = q4[2 * j], qb = q4[2 * j + 1];
    float4 ka = k4[2 * j], kb = k4[2 * j + 1];
    h8 ev = e8[j];
    sum += qa.x * (ka.x + (float)ev[0]) + qa.y * (ka.y + (float)ev[1]) +
           qa.z * (ka.z + (float)ev[2]) + qa.w * (ka.w + (float)ev[3]) +
           qb.x * (kb.x + (float)ev[4]) + qb.y * (kb.y + (float)ev[5]) +
           qb.z * (kb.z + (float)ev[6]) + qb.w * (kb.w + (float)ev[7]);
  }
  sum *= 0.1767766953f;  // 1/sqrt(32)
  alpha[e] = sum;
  atomicMax(amaxU + d, fenc(sum));
}

__global__ void attn_pass2(const int* __restrict__ dst,
                           float* __restrict__ alpha,
                           const unsigned int* __restrict__ amaxU,
                           float* __restrict__ denom, float* __restrict__ deg,
                           int E) {
  int e = blockIdx.x * blockDim.x + threadIdx.x;
  if (e >= E) return;
  int d = dst[e];
  float ex = __expf(alpha[e] - fdec(amaxU[d]));
  alpha[e] = ex;
  atomicAdd(denom + d, ex);
  atomicAdd(deg + d, 1.0f);
}

__global__ void attn_pass3(const int* __restrict__ src,
                           const int* __restrict__ dst,
                           const float* __restrict__ alpha,
                           const float* __restrict__ denom,
                           const float* __restrict__ v,
                           const _Float16* __restrict__ ep,
                           float* __restrict__ agg, int E) {
  int e = blockIdx.x * blockDim.x + threadIdx.x;
  if (e >= E) return;
  int s = src[e], d = dst[e];
  float a = alpha[e] / denom[d];
  const float4* v4 = (const float4*)(v + (size_t)s * 32);
  const h8* e8 = (const h8*)(ep + (size_t)e * 32);
  float* ag = agg + (size_t)d * 32;
#pragma unroll
  for (int j = 0; j < 4; ++j) {
    float4 va = v4[2 * j], vb = v4[2 * j + 1];
    h8 ev = e8[j];
    atomicAdd(ag + 8 * j + 0, (va.x + (float)ev[0]) * a);
    atomicAdd(ag + 8 * j + 1, (va.y + (float)ev[1]) * a);
    atomicAdd(ag + 8 * j + 2, (va.z + (float)ev[2]) * a);
    atomicAdd(ag + 8 * j + 3, (va.w + (float)ev[3]) * a);
    atomicAdd(ag + 8 * j + 4, (vb.x + (float)ev[4]) * a);
    atomicAdd(ag + 8 * j + 5, (vb.y + (float)ev[5]) * a);
    atomicAdd(ag + 8 * j + 6, (vb.z + (float)ev[6]) * a);
    atomicAdd(ag + 8 * j + 7, (vb.w + (float)ev[7]) * a);
  }
}

// out = relu(agg/max(deg,1) + X@Wskip + bskip); also refresh f16 comb rows.
__global__ __launch_bounds__(256) void finalize_nodes(
    const _Float16* __restrict__ X, const _Float16* __restrict__ WskT,
    const float* __restrict__ bsk, const float* __restrict__ agg,
    const float* __restrict__ deg, float* __restrict__ outDst,
    _Float16* __restrict__ Xout, int M) {
  int wave = blockIdx.x * (blockDim.x >> 5) + (threadIdx.x >> 5);
  int base = wave * 16;
  if (base >= M) return;
  int l = threadIdx.x & 31;
  int g = l >> 4, col = l & 15;
  const _Float16* arow = X + (size_t)(base + col) * KP;
  v8f acc0 = {}, acc1 = {};
#pragma unroll
  for (int c = 0; c < 2; ++c) {
    int ka = c * 32;
    v16h a = mkfrag(ldh8(arow + ka + g * 16), ldh8(arow + ka + g * 16 + 8));
    const _Float16* w0 = WskT + (size_t)col * KP + ka + g * 16;
    const _Float16* w1 = WskT + (size_t)(col + 16) * KP + ka + g * 16;
    acc0 = wmma_f16(a, mkfrag(ldh8(w0), ldh8(w0 + 8)), acc0);
    acc1 = wmma_f16(a, mkfrag(ldh8(w1), ldh8(w1 + 8)), acc1);
  }
  float b0 = bsk[col], b1 = bsk[col + 16];
#pragma unroll
  for (int r = 0; r < 8; ++r) {
    int row = base + r + 8 * g;
    float inv = 1.f / fmaxf(deg[row], 1.f);
    float o0 = fmaxf(agg[(size_t)row * 32 + col] * inv + acc0[r] + b0, 0.f);
    float o1 = fmaxf(agg[(size_t)row * 32 + col + 16] * inv + acc1[r] + b1, 0.f);
    outDst[(size_t)row * 32 + col] = o0;
    outDst[(size_t)row * 32 + col + 16] = o1;
    Xout[(size_t)row * KP + col] = (_Float16)o0;        // lp cols 32..39 untouched
    Xout[(size_t)row * KP + col + 16] = (_Float16)o1;
  }
}

// ------------------------ fused per-edge 2-layer MLP -----------------------
// x = [edge_comb(40) | vq[vidx](32) | cq[cidx](32)] padded to 128.
__global__ __launch_bounds__(256) void edge_mlp(
    const _Float16* __restrict__ edgeF16, const _Float16* __restrict__ vqF16,
    const _Float16* __restrict__ cqF16, const int* __restrict__ eiv,
    const int* __restrict__ eic, const _Float16* __restrict__ eW1T,
    const float* __restrict__ eb1, const _Float16* __restrict__ eW2T,
    const float* __restrict__ eb2, float* __restrict__ outE, int E) {
  __shared__ _Float16 smem[8 * 512];
  int wave = blockIdx.x * 8 + (threadIdx.x >> 5);
  int base = wave * 16;
  if (base >= E) return;
  _Float16* sm = smem + (threadIdx.x >> 5) * 512;
  int l = threadIdx.x & 31;
  int g = l >> 4, col = l & 15;
  int e = base + col;
  int vidx = eiv[e], cidx = eic[e];
  const _Float16* erow = edgeF16 + (size_t)e * KP;
  const _Float16* vrow = vqF16 + (size_t)vidx * 32;
  const _Float16* crow = cqF16 + (size_t)cidx * 32;

  v8f acc0 = {}, acc1 = {};
  h8 alos[4], ahis[4];
  alos[0] = ldh8(erow + g * 16);       ahis[0] = ldh8(erow + 8 + g * 16);
  alos[1] = ldh8(erow + 32 + g * 16);  ahis[1] = ldh8(erow + 40 + g * 16);
  alos[2] = ldh8(vrow + g * 16);       ahis[2] = ldh8(vrow + 8 + g * 16);
  alos[3] = ldh8(crow + g * 16);       ahis[3] = ldh8(crow + 8 + g * 16);
#pragma unroll
  for (int c = 0; c < 4; ++c) {
    int ko = c * 32;
    v16h a = mkfrag(alos[c], ahis[c]);
    const _Float16* w0 = eW1T + (size_t)col * 128 + ko + g * 16;
    const _Float16* w1 = eW1T + (size_t)(col + 16) * 128 + ko + g * 16;
    acc0 = wmma_f16(a, mkfrag(ldh8(w0), ldh8(w0 + 8)), acc0);
    acc1 = wmma_f16(a, mkfrag(ldh8(w1), ldh8(w1 + 8)), acc1);
  }
  float b10 = eb1[col], b11 = eb1[col + 16];
#pragma unroll
  for (int r = 0; r < 8; ++r) {
    int row = r + 8 * g;
    sm[row * 32 + col] = (_Float16)fmaxf(acc0[r] + b10, 0.f);
    sm[row * 32 + col + 16] = (_Float16)fmaxf(acc1[r] + b11, 0.f);
  }
  // in-wave LDS re-shape: C layout -> A layout (DS ops in-order per wave)
  v16h a2 = mkfrag(ldh8(sm + col * 32 + g * 16), ldh8(sm + col * 32 + 8 + g * 16));
  const _Float16* w0 = eW2T + (size_t)col * 32 + g * 16;
  const _Float16* w1 = eW2T + (size_t)(col + 16) * 32 + g * 16;
  v8f o0 = {}, o1 = {};
  o0 = wmma_f16(a2, mkfrag(ldh8(w0), ldh8(w0 + 8)), o0);
  o1 = wmma_f16(a2, mkfrag(ldh8(w1), ldh8(w1 + 8)), o1);
  float b20 = eb2[col], b21 = eb2[col + 16];
#pragma unroll
  for (int r = 0; r < 8; ++r) {
    int row = base + r + 8 * g;
    outE[(size_t)row * 32 + col] = fmaxf(o0[r] + b20, 0.f);
    outE[(size_t)row * 32 + col + 16] = fmaxf(o1[r] + b21, 0.f);
  }
}

// ------------------------------- host side ---------------------------------

static inline int cdiv(int a, int b) { return (a + b - 1) / b; }

static void run_conv(hipStream_t stream, const _Float16* srcF16, int Msrc,
                     _Float16* dstF16, int Mdst, const int* srcIdx,
                     const int* dstIdx, const _Float16* edgeF16,
                     _Float16* eprojF16, const _Float16* WqT, const float* bq,
                     const _Float16* WkT, const float* bk, const _Float16* WvT,
                     const float* bv, const _Float16* WeT,
                     const _Float16* WskT, const float* bsk, float* qbuf,
                     float* kbuf, float* vbuf, float* agg,
                     unsigned int* amaxU, float* denom, float* deg,
                     float* alpha, float* outDst) {
  // q (dst), k/v (src) projections
  gemm_rows<2, false, false><<<cdiv(Mdst, 128), 256, 0, stream>>>(
      dstF16, KP, WqT, KP, bq, qbuf, nullptr, Mdst);
  gemm_rows<2, false, false><<<cdiv(Msrc, 128), 256, 0, stream>>>(
      srcF16, KP, WkT, KP, bk, kbuf, nullptr, Msrc);
  gemm_rows<2, false, false><<<cdiv(Msrc, 128), 256, 0, stream>>>(
      srcF16, KP, WvT, KP, bv, vbuf, nullptr, Msrc);
  // e = edge_comb @ We (no bias), stored f16
  gemm_rows<2, false, true><<<cdiv(EN, 128), 256, 0, stream>>>(
      edgeF16, KP, WeT, KP, nullptr, nullptr, eprojF16, EN);
  init_node_state<<<cdiv(Mdst, 256), 256, 0, stream>>>(amaxU, denom, deg, Mdst);
  zero_f32<<<cdiv(Mdst * 32, 256), 256, 0, stream>>>(agg, Mdst * 32);
  attn_pass1<<<cdiv(EN, 256), 256, 0, stream>>>(srcIdx, dstIdx, qbuf, kbuf,
                                                eprojF16, alpha, amaxU, EN);
  attn_pass2<<<cdiv(EN, 256), 256, 0, stream>>>(dstIdx, alpha, amaxU, denom,
                                                deg, EN);
  attn_pass3<<<cdiv(EN, 256), 256, 0, stream>>>(srcIdx, dstIdx, alpha, denom,
                                                vbuf, eprojF16, agg, EN);
  finalize_nodes<<<cdiv(Mdst, 128), 256, 0, stream>>>(dstF16, WskT, bsk, agg,
                                                      deg, outDst, dstF16,
                                                      Mdst);
}

extern "C" void kernel_launch(void* const* d_in, const int* in_sizes, int n_in,
                              void* d_out, int out_size, void* d_ws,
                              size_t ws_size, hipStream_t stream) {
  (void)in_sizes; (void)n_in; (void)out_size; (void)ws_size;
  const float* var_l = (const float*)d_in[0];
  const float* var_lp = (const float*)d_in[1];
  const float* con_l = (const float*)d_in[2];
  const float* con_lp = (const float*)d_in[3];
  const float* edge_l = (const float*)d_in[4];
  const float* lo = (const float*)d_in[5];
  const float* hi = (const float*)d_in[6];
  const float* dm = (const float*)d_in[7];
  const float* elp = (const float*)d_in[8];
  const int* eiv = (const int*)d_in[9];
  const int* eic = (const int*)d_in[10];
  float* O = (float*)d_out;

  // workspace carve
  uint8_t* w = (uint8_t*)d_ws;
  size_t off = 0;
  auto alloc = [&](size_t bytes) -> void* {
    off = (off + 255) & ~(size_t)255;
    void* p = w + off;
    off += bytes;
    return p;
  };
  _Float16* varF16 = (_Float16*)alloc((size_t)NVN * KP * 2);
  _Float16* conF16 = (_Float16*)alloc((size_t)NCN * KP * 2);
  _Float16* edgeF16 = (_Float16*)alloc((size_t)EN * KP * 2);
  _Float16* eprojF16 = (_Float16*)alloc((size_t)EN * 32 * 2);
  float* alpha = (float*)alloc((size_t)EN * 4);
  float* qbuf = (float*)alloc((size_t)NVN * 32 * 4);
  float* kbuf = (float*)alloc((size_t)NVN * 32 * 4);
  float* vbuf = (float*)alloc((size_t)NVN * 32 * 4);
  float* agg = (float*)alloc((size_t)NVN * 32 * 4);
  unsigned int* amaxU = (unsigned int*)alloc((size_t)NVN * 4);
  float* denom = (float*)alloc((size_t)NVN * 4);
  float* deg = (float*)alloc((size_t)NVN * 4);
  _Float16* hF16 = (_Float16*)alloc((size_t)NVN * 32 * 2);
  _Float16* vqF16 = (_Float16*)alloc((size_t)NVN * 32 * 2);
  _Float16* cqF16 = (_Float16*)alloc((size_t)NCN * 32 * 2);
  _Float16* WT[10];  // con {q,k,v,e,skip}, var {q,k,v,e,skip}, each [32][64]
  for (int i = 0; i < 10; ++i) WT[i] = (_Float16*)alloc(32 * KP * 2);
  _Float16* vW1T = (_Float16*)alloc(32 * 64 * 2);
  _Float16* vW2T = (_Float16*)alloc(32 * 32 * 2);
  _Float16* cW1T = (_Float16*)alloc(32 * 64 * 2);
  _Float16* cW2T = (_Float16*)alloc(32 * 32 * 2);
  _Float16* eW1T = (_Float16*)alloc(32 * 128 * 2);
  _Float16* eW2T = (_Float16*)alloc(32 * 32 * 2);

  // pack features to f16 (K padded to 64)
  pack_nodes<<<cdiv(NVN, 256), 256, 0, stream>>>(var_l, var_lp, varF16, NVN);
  pack_nodes<<<cdiv(NCN, 256), 256, 0, stream>>>(con_l, con_lp, conF16, NCN);
  pack_edges<<<cdiv(EN, 256), 256, 0, stream>>>(edge_l, lo, hi, dm, elp,
                                                edgeF16, EN);

  // convert weights to transposed/padded f16
  auto cw = [&](int idx, _Float16* dst, int K, int Kpad, int remap) {
    convert_weight<<<cdiv(32 * Kpad, 256), 256, 0, stream>>>(
        (const float*)d_in[idx], dst, K, Kpad, remap);
  };
  // con_conv: 11 Wq 13 Wk 15 Wv 17 We 18 Wskip; var_conv: +9
  cw(11, WT[0], 40, KP, 0); cw(13, WT[1], 40, KP, 0); cw(15, WT[2], 40, KP, 0);
  cw(17, WT[3], 40, KP, 0); cw(18, WT[4], 40, KP, 0);
  cw(20, WT[5], 40, KP, 0); cw(22, WT[6], 40, KP, 0); cw(24, WT[7], 40, KP, 0);
  cw(26, WT[8], 40, KP, 0); cw(27, WT[9], 40, KP, 0);
  cw(29, vW1T, 40, 64, 0); cw(31, vW2T, 32, 32, 0);
  cw(33, cW1T, 40, 64, 0); cw(35, cW2T, 32, 32, 0);
  cw(37, eW1T, 104, 128, 1); cw(39, eW2T, 32, 32, 0);

  const float* bq_c = (const float*)d_in[12];
  const float* bk_c = (const float*)d_in[14];
  const float* bv_c = (const float*)d_in[16];
  const float* bsk_c = (const float*)d_in[19];
  const float* bq_v = (const float*)d_in[21];
  const float* bk_v = (const float*)d_in[23];
  const float* bv_v = (const float*)d_in[25];
  const float* bsk_v = (const float*)d_in[28];

  // conv1: var -> con ; output con_new at O + NV*32 ; refreshes conF16
  run_conv(stream, varF16, NVN, conF16, NCN, eiv, eic, edgeF16, eprojF16,
           WT[0], bq_c, WT[1], bk_c, WT[2], bv_c, WT[3], WT[4], bsk_c, qbuf,
           kbuf, vbuf, agg, amaxU, denom, deg, alpha, O + (size_t)NVN * 32);
  // conv2: con(new) -> var ; output var_new at O ; refreshes varF16
  run_conv(stream, conF16, NCN, varF16, NVN, eic, eiv, edgeF16, eprojF16,
           WT[5], bq_v, WT[6], bk_v, WT[7], bv_v, WT[8], WT[9], bsk_v, qbuf,
           kbuf, vbuf, agg, amaxU, denom, deg, alpha, O);

  // node compression MLPs (relu both layers), f16 outputs for gathering
  const float* vb1 = (const float*)d_in[30];
  const float* vb2 = (const float*)d_in[32];
  const float* cb1 = (const float*)d_in[34];
  const float* cb2 = (const float*)d_in[36];
  const float* eb1 = (const float*)d_in[38];
  const float* eb2 = (const float*)d_in[40];
  gemm_rows<2, true, true><<<cdiv(NVN, 128), 256, 0, stream>>>(
      varF16, KP, vW1T, 64, vb1, nullptr, hF16, NVN);
  gemm_rows<1, true, true><<<cdiv(NVN, 128), 256, 0, stream>>>(
      hF16, 32, vW2T, 32, vb2, nullptr, vqF16, NVN);
  gemm_rows<2, true, true><<<cdiv(NCN, 128), 256, 0, stream>>>(
      conF16, KP, cW1T, 64, cb1, nullptr, hF16, NCN);
  gemm_rows<1, true, true><<<cdiv(NCN, 128), 256, 0, stream>>>(
      hF16, 32, cW2T, 32, cb2, nullptr, cqF16, NCN);

  // fused per-edge 2-layer MLP -> edge_new at O + (NV+NC)*32
  edge_mlp<<<cdiv(EN / 16, 8), 256, 0, stream>>>(
      edgeF16, vqF16, cqF16, eiv, eic, eW1T, eb1, eW2T, eb2,
      O + (size_t)(NVN + NCN) * 32, EN);
}
